// MultiHeadAttention_72310069395731
// MI455X (gfx1250) — compile-verified, matrix operand order checked by reading, not run on hardware
//
#include <hip/hip_runtime.h>
#include <hip/hip_bf16.h>

#define D_MODEL 1024
#define NUM_HEADS 16
#define D_K 64
#define SEQ 2048
#define BATCH 2

typedef __attribute__((ext_vector_type(16))) _Float16 v16h;
typedef __attribute__((ext_vector_type(4)))  _Float16 v4h;
typedef __attribute__((ext_vector_type(8)))  float    v8f;

// Per-lane fragment packing for V_WMMA_F32_16X16X32_F16 (ISA 7.12.2):
// lanes 0-15: K = kbase+0..7 (elems 0..7), kbase+16..23 (elems 8..15)
// lanes 16-31: +8 on the K base. Same pattern for A (lane=M row) and B (lane=N col).
__device__ __forceinline__ v16h frag_from_f16(const _Float16* __restrict__ row, int kbase, int laneHi) {
  const _Float16* p = row + kbase + (laneHi << 3);
  v16h f;
#pragma unroll
  for (int j = 0; j < 8; ++j) {
    f[j]     = p[j];
    f[j + 8] = p[j + 16];
  }
  return f;
}

__device__ __forceinline__ v16h frag_from_lds(const _Float16* row, int kbase, int laneHi) {
  const _Float16* p = row + kbase + (laneHi << 3);
  v16h f;
#pragma unroll
  for (int j = 0; j < 8; ++j) {
    f[j]     = p[j];
    f[j + 8] = p[j + 16];
  }
  return f;
}

__device__ __forceinline__ v8f wmma_f16(v16h a, v16h b, v8f c) {
  return __builtin_amdgcn_wmma_f32_16x16x32_f16(
      /*neg_a=*/false, a, /*neg_b=*/false, b,
      /*c_mod=*/(short)0, c, /*reuse_a=*/false, /*reuse_b=*/false);
}

// ---------------------------------------------------------------------------
// Kernel 0: one-shot f32 -> f16 conversion (4 elements / thread, vectorized).
// ---------------------------------------------------------------------------
__global__ void __launch_bounds__(256)
cvt_f32_f16_kernel(const float* __restrict__ src, _Float16* __restrict__ dst) {
  const size_t i = ((size_t)blockIdx.x * 256 + threadIdx.x) * 4;
  const float4 v = *(const float4*)(src + i);
  v4h h;
  h[0] = (_Float16)v.x; h[1] = (_Float16)v.y;
  h[2] = (_Float16)v.z; h[3] = (_Float16)v.w;
  *(v4h*)(dst + i) = h;
}

// ---------------------------------------------------------------------------
// Kernel 1: Q/K/V projection  y = x @ W^T  (z selects which projection).
// Each wave: 16(M) x 64(N) tile, 4 independent accums, A frag reused 4x.
// Q,K -> [B,H,S,64] f16 ; V -> transposed [B,H,64,S] f16.
// ---------------------------------------------------------------------------
__global__ void __launch_bounds__(128)
qkv_proj_kernel(const _Float16* __restrict__ Xq, const _Float16* __restrict__ Xk,
                const _Float16* __restrict__ Xv,
                const _Float16* __restrict__ Wq, const _Float16* __restrict__ Wk,
                const _Float16* __restrict__ Wv,
                _Float16* __restrict__ Qo, _Float16* __restrict__ Ko,
                _Float16* __restrict__ Vt) {
  const int lane   = threadIdx.x & 31;
  const int laneLo = lane & 15, laneHi = lane >> 4;
  const int w  = threadIdx.x >> 5;                 // wave id 0..3
  const int m0 = blockIdx.x * 16;                  // row tile over B*S
  const int n0 = (blockIdx.y * 4 + w) * 64;        // wave's 64-wide N tile (== one head)
  const int z  = blockIdx.z;                       // 0=Q 1=K 2=V

  const _Float16* X = (z == 0) ? Xq : (z == 1) ? Xk : Xv;
  const _Float16* W = (z == 0) ? Wq : (z == 1) ? Wk : Wv;

  const _Float16* rowA  = X + (size_t)(m0 + laneLo) * D_MODEL;
  const _Float16* rowB0 = W + (size_t)(n0 + laneLo) * D_MODEL;   // B[k,n] = W[n,k]

  v8f zero = {};
  v8f acc[4];
#pragma unroll
  for (int j = 0; j < 4; ++j) acc[j] = zero;

#pragma unroll 2
  for (int kk = 0; kk < D_MODEL; kk += 32) {
    const v16h a = frag_from_f16(rowA, kk, laneHi);
#pragma unroll
    for (int j = 0; j < 4; ++j) {
      v16h b = frag_from_f16(rowB0 + (size_t)(j * 16) * D_MODEL, kk, laneHi);
      acc[j] = wmma_f16(a, b, acc[j]);
    }
  }

#pragma unroll
  for (int j = 0; j < 4; ++j) {
    const int n = n0 + j * 16 + laneLo;
    const int h = n >> 6, d = n & 63;
#pragma unroll
    for (int r = 0; r < 8; ++r) {
      const int m  = m0 + laneHi * 8 + r;          // C layout: M = r + 8*laneHi
      const int bb = m >> 11, s = m & 2047;
      const size_t bh = (size_t)bb * NUM_HEADS + h;
      const _Float16 val = (_Float16)acc[j][r];
      if (z == 0)      Qo[(bh * SEQ + s) * D_K + d] = val;
      else if (z == 1) Ko[(bh * SEQ + s) * D_K + d] = val;
      else             Vt[(bh * D_K + d) * SEQ + s] = val;
    }
  }
}

// ---------------------------------------------------------------------------
// Kernel 2 (fused): scores = (Q K^T)/8 -> mask -> softmax -> write attn once
// -> context = attn @ V, all in one block per (b, h, 16 q-rows).
// Stage 1: 8 waves, each holds a 16x256 normalized strip in 16 v8f accums.
// Strip is also dumped as f16 into a 64KB LDS tile (cross-lane relayout).
// Stage 2: waves split K=2048 (256 keys each): A-frags from LDS, B-frags from
// Vt[B,H,64,S]; 8 partial 16x64 tiles reduced via LDS (reusing the strip).
// Dynamic LDS: 64KB.
// ---------------------------------------------------------------------------
__global__ void __launch_bounds__(256)
attn_softmax_ctx_kernel(const _Float16* __restrict__ Qo, const _Float16* __restrict__ Ko,
                        const _Float16* __restrict__ Vt, const int* __restrict__ mask,
                        float* __restrict__ attn, _Float16* __restrict__ ctx) {
  extern __shared__ __align__(16) char smem_raw[];
  _Float16* wLDS   = (_Float16*)smem_raw;          // [16][2048] f16 = 64KB
  float*    partial = (float*)smem_raw;            // [8][16][64] f32 = 32KB (reused)
  __shared__ float redmax[128], rowmax[16], redsum[128], rowsum[16];

  const int lane   = threadIdx.x & 31;
  const int laneLo = lane & 15, laneHi = lane >> 4;
  const int w  = threadIdx.x >> 5;          // wave id 0..7 -> keys [w*256, w*256+256)
  const int q0 = blockIdx.x * 16;
  const int h  = blockIdx.y, b = blockIdx.z;
  const size_t bh = (size_t)b * NUM_HEADS + h;

  const _Float16* Qbase = Qo + (bh * SEQ + q0) * D_K;
  const _Float16* Kbase = Ko + bh * SEQ * D_K;

  const v16h a0 = frag_from_f16(Qbase + (size_t)laneLo * D_K, 0,  laneHi);
  const v16h a1 = frag_from_f16(Qbase + (size_t)laneLo * D_K, 32, laneHi);

  v8f zero = {};
  v8f acc[16];
#pragma unroll
  for (int t = 0; t < 16; ++t) acc[t] = zero;

#pragma unroll
  for (int t = 0; t < 16; ++t) {
    const _Float16* rowB = Kbase + (size_t)(w * 256 + t * 16 + laneLo) * D_K;
    v16h b0 = frag_from_f16(rowB, 0,  laneHi);
    v16h b1 = frag_from_f16(rowB, 32, laneHi);
    acc[t] = wmma_f16(a0, b0, acc[t]);
    acc[t] = wmma_f16(a1, b1, acc[t]);
  }

  // scale + mask + per-lane row maxima
  const float NEG_INF = -__builtin_inff();
  float pmax[8];
#pragma unroll
  for (int r = 0; r < 8; ++r) pmax[r] = NEG_INF;
#pragma unroll
  for (int t = 0; t < 16; ++t) {
    const int col = w * 256 + t * 16 + laneLo;
    const bool dead = (mask[b * SEQ + col] == 0);
#pragma unroll
    for (int r = 0; r < 8; ++r) {
      float sc = acc[t][r] * 0.125f;        // / sqrt(64)
      if (dead) sc = NEG_INF;
      acc[t][r] = sc;
      pmax[r] = fmaxf(pmax[r], sc);
    }
  }
  // rows split lanes0-15 (M0..7) / lanes16-31 (M8..15): xor-reduce over 16 lanes
#pragma unroll
  for (int off = 1; off < 16; off <<= 1) {
#pragma unroll
    for (int r = 0; r < 8; ++r) pmax[r] = fmaxf(pmax[r], __shfl_xor(pmax[r], off, 32));
  }
  if (laneLo == 0) {
#pragma unroll
    for (int r = 0; r < 8; ++r) redmax[(laneHi * 8 + r) * 8 + w] = pmax[r];
  }
  __syncthreads();
  if (threadIdx.x < 16) {
    float m = NEG_INF;
    for (int j = 0; j < 8; ++j) m = fmaxf(m, redmax[threadIdx.x * 8 + j]);
    rowmax[threadIdx.x] = m;
  }
  __syncthreads();

  float psum[8];
#pragma unroll
  for (int r = 0; r < 8; ++r) psum[r] = 0.0f;
#pragma unroll
  for (int t = 0; t < 16; ++t) {
#pragma unroll
    for (int r = 0; r < 8; ++r) {
      float e = __expf(acc[t][r] - rowmax[laneHi * 8 + r]);
      acc[t][r] = e;
      psum[r] += e;
    }
  }
#pragma unroll
  for (int off = 1; off < 16; off <<= 1) {
#pragma unroll
    for (int r = 0; r < 8; ++r) psum[r] += __shfl_xor(psum[r], off, 32);
  }
  if (laneLo == 0) {
#pragma unroll
    for (int r = 0; r < 8; ++r) redsum[(laneHi * 8 + r) * 8 + w] = psum[r];
  }
  __syncthreads();
  if (threadIdx.x < 16) {
    float s = 0.0f;
    for (int j = 0; j < 8; ++j) s += redsum[threadIdx.x * 8 + j];
    rowsum[threadIdx.x] = s;
  }
  __syncthreads();

  // normalize: write attn (f32, global, once) + weight strip (f16, LDS)
#pragma unroll
  for (int r = 0; r < 8; ++r) {
    const int row = laneHi * 8 + r;
    const float inv = 1.0f / rowsum[row];
    float* outrow = attn + (bh * SEQ + q0 + row) * SEQ;
#pragma unroll
    for (int t = 0; t < 16; ++t) {
      const int col = w * 256 + t * 16 + laneLo;
      const float p = acc[t][r] * inv;
      outrow[col] = p;
      wLDS[row * SEQ + col] = (_Float16)p;
    }
  }
  __syncthreads();

  // Stage 2: context partial over this wave's 256 keys (8 WMMA k-steps).
  const _Float16* rowB0 = Vt + (bh * D_K + laneLo) * SEQ;        // B[k,n]=V[k,d]
  const _Float16* ldsRowA = wLDS + laneLo * SEQ;
  v8f cacc[4];
#pragma unroll
  for (int j = 0; j < 4; ++j) cacc[j] = zero;
#pragma unroll
  for (int s8 = 0; s8 < 8; ++s8) {
    const int kk = w * 256 + s8 * 32;
    const v16h a = frag_from_lds(ldsRowA, kk, laneHi);
#pragma unroll
    for (int j = 0; j < 4; ++j) {
      v16h bfr = frag_from_f16(rowB0 + (size_t)(j * 16) * SEQ, kk, laneHi);
      cacc[j] = wmma_f16(a, bfr, cacc[j]);
    }
  }
  __syncthreads();   // everyone done reading wLDS; safe to reuse as partial[]

#pragma unroll
  for (int j = 0; j < 4; ++j) {
#pragma unroll
    for (int r = 0; r < 8; ++r) {
      const int row = laneHi * 8 + r;
      const int col = j * 16 + laneLo;
      partial[(w * 16 + row) * 64 + col] = cacc[j][r];
    }
  }
  __syncthreads();

  // reduce 8 partials -> ctx (f16). 1024 outputs, 4 per thread.
#pragma unroll
  for (int i = 0; i < 4; ++i) {
    const int o   = threadIdx.x + i * 256;
    const int row = o >> 6, col = o & 63;
    float s = 0.0f;
#pragma unroll
    for (int w2 = 0; w2 < 8; ++w2) s += partial[(w2 * 16 + row) * 64 + col];
    ctx[((size_t)b * SEQ + q0 + row) * D_MODEL + h * D_K + col] = (_Float16)s;
  }
}

// ---------------------------------------------------------------------------
// Kernel 3: output = ctx @ W_O^T -> f32 d_out. 16x64-per-wave tiling.
// ---------------------------------------------------------------------------
__global__ void __launch_bounds__(128)
out_proj_kernel(const _Float16* __restrict__ ctx, const _Float16* __restrict__ Wo,
                float* __restrict__ out) {
  const int lane   = threadIdx.x & 31;
  const int laneLo = lane & 15, laneHi = lane >> 4;
  const int w  = threadIdx.x >> 5;
  const int m0 = blockIdx.x * 16;
  const int n0 = (blockIdx.y * 4 + w) * 64;

  const _Float16* rowA  = ctx + (size_t)(m0 + laneLo) * D_MODEL;
  const _Float16* rowB0 = Wo  + (size_t)(n0 + laneLo) * D_MODEL;

  v8f zero = {};
  v8f acc[4];
#pragma unroll
  for (int j = 0; j < 4; ++j) acc[j] = zero;

#pragma unroll 2
  for (int kk = 0; kk < D_MODEL; kk += 32) {
    const v16h a = frag_from_f16(rowA, kk, laneHi);
#pragma unroll
    for (int j = 0; j < 4; ++j) {
      v16h b = frag_from_f16(rowB0 + (size_t)(j * 16) * D_MODEL, kk, laneHi);
      acc[j] = wmma_f16(a, b, acc[j]);
    }
  }

#pragma unroll
  for (int j = 0; j < 4; ++j) {
#pragma unroll
    for (int r = 0; r < 8; ++r)
      out[(size_t)(m0 + laneHi * 8 + r) * D_MODEL + n0 + j * 16 + laneLo] = acc[j][r];
  }
}

extern "C" void kernel_launch(void* const* d_in, const int* in_sizes, int n_in,
                              void* d_out, int out_size, void* d_ws, size_t ws_size,
                              hipStream_t stream) {
  const float* q    = (const float*)d_in[0];
  const float* k    = (const float*)d_in[1];
  const float* v    = (const float*)d_in[2];
  const int*   mask = (const int*)d_in[3];
  const float* wq   = (const float*)d_in[4];
  const float* wk   = (const float*)d_in[5];
  const float* wv   = (const float*)d_in[6];
  const float* wo   = (const float*)d_in[7];

  float* out  = (float*)d_out;                                  // [B,S,D]
  float* attn = (float*)d_out + (size_t)BATCH * SEQ * D_MODEL;  // [B,H,S,S]

  _Float16* ws = (_Float16*)d_ws;
  const size_t NX = (size_t)BATCH * SEQ * D_MODEL;              // 4,194,304
  const size_t NW = (size_t)D_MODEL * D_MODEL;                  // 1,048,576
  _Float16* Qo  = ws;                 // [B,H,S,64]
  _Float16* Ko  = Qo  + NX;
  _Float16* Vt  = Ko  + NX;           // [B,H,64,S]
  _Float16* ctx = Vt  + NX;           // [B,S,D]
  _Float16* Xq  = ctx + NX;
  _Float16* Xk  = Xq  + NX;
  _Float16* Xv  = Xk  + NX;
  _Float16* Wqh = Xv  + NX;
  _Float16* Wkh = Wqh + NW;
  _Float16* Wvh = Wkh + NW;
  _Float16* Woh = Wvh + NW;

  // one-shot f32 -> f16 conversions
  cvt_f32_f16_kernel<<<NX / 1024, 256, 0, stream>>>(q,  Xq);
  cvt_f32_f16_kernel<<<NX / 1024, 256, 0, stream>>>(k,  Xk);
  cvt_f32_f16_kernel<<<NX / 1024, 256, 0, stream>>>(v,  Xv);
  cvt_f32_f16_kernel<<<NW / 1024, 256, 0, stream>>>(wq, Wqh);
  cvt_f32_f16_kernel<<<NW / 1024, 256, 0, stream>>>(wk, Wkh);
  cvt_f32_f16_kernel<<<NW / 1024, 256, 0, stream>>>(wv, Wvh);
  cvt_f32_f16_kernel<<<NW / 1024, 256, 0, stream>>>(wo, Woh);

  qkv_proj_kernel<<<dim3(BATCH * SEQ / 16, 4, 3), 128, 0, stream>>>(
      Xq, Xk, Xv, Wqh, Wkh, Wvh, Qo, Ko, Vt);

  const size_t smem_bytes = (size_t)16 * SEQ * sizeof(_Float16);   // 64KB strip
  attn_softmax_ctx_kernel<<<dim3(SEQ / 16, NUM_HEADS, BATCH), 256, smem_bytes, stream>>>(
      Qo, Ko, Vt, mask, attn, ctx);

  out_proj_kernel<<<dim3(BATCH * SEQ / 16, 4), 128, 0, stream>>>(
      ctx, Woh, out);
}
